// Custom_Model_59047210385498
// MI455X (gfx1250) — compile-verified, matrix operand order checked by reading, not run on hardware
//
#include <hip/hip_runtime.h>

#define EDGES 131072
#define TRIP  1048576
#define HC    128
#define INTC  64
#define NRAD  6
#define NSPH  7
#define BASIS 8

typedef __bf16 v16b __attribute__((ext_vector_type(16)));
typedef float  v8f  __attribute__((ext_vector_type(8)));

__device__ __forceinline__ float silu_f(float v) {
    return v / (1.0f + __expf(-v));
}

// ---------------------------------------------------------------------------
// WMMA GEMM: out[M,N] = epi(A[M,K] @ W[K,N]) with epi = (+bias) -> silu -> *mul -> +addv
// A, W fp32 in global, converted to bf16 in registers (v_cvt_pk_bf16_f32).
// v_wmma_f32_16x16x32_bf16; block = 256 threads (8 wave32 waves).
// Wave tile: 32 rows x N/2 cols (2 row-tiles x N/32 n-tiles). Waves are
// (rowgroup 0..3) x (n-half 0..1); block covers 128 rows. Grid.x = M/128.
// Per k-step: 2 A fragments + NTW B fragments loaded first, then the
// 2*NTW WMMAs chain with each B fragment reused twice (halves DS traffic
// per WMMA and keeps the fragment working set at ~112 VGPRs so the
// scheduler can hoist all loads ahead of the WMMA chain).
// ---------------------------------------------------------------------------
template<int K, int N, bool BIAS, bool MUL, bool ADD>
__global__ __launch_bounds__(256) void wmma_gemm_kernel(
    const float* __restrict__ A, const float* __restrict__ W,
    const float* __restrict__ bias, const float* __restrict__ mul,
    const float* __restrict__ addv, float* __restrict__ out)
{
    constexpr int NTW  = N / 32;     // n-tiles per wave (4 for N=128, 2 for N=64)
    constexpr int Kpad = K + 8;
    __shared__ __bf16 ldsW[N * Kpad];
    const int tid = threadIdx.x;

    // Stage W (row-major [K][N], fp32) transposed into LDS: Wt[n][k] (bf16)
    for (int i = tid; i < K * N; i += 256) {
        int k = i / N, n = i - k * N;
        ldsW[n * Kpad + k] = (__bf16)W[i];
    }
    __syncthreads();

    const int wave   = tid >> 5;
    const int lane   = tid & 31;
    const int lhalf  = lane >> 4;      // 0: lanes 0-15, 1: lanes 16-31
    const int l16    = lane & 15;
    const int rowgrp = wave >> 1;      // 0..3
    const int nbase  = (wave & 1) * (N / 2);
    const size_t row0 = (size_t)blockIdx.x * 128 + rowgrp * 32 + l16;

    v8f acc[2][NTW] = {};

    #pragma unroll
    for (int k0 = 0; k0 < K; k0 += 32) {
        // Two A fragments (rows row0, row0+16); lane holds 16 contiguous K values
        v16b a[2];
        #pragma unroll
        for (int mt = 0; mt < 2; ++mt) {
            const float* arow = A + (row0 + mt * 16) * (size_t)K + k0 + (lhalf << 4);
            #pragma unroll
            for (int i = 0; i < 16; ++i) a[mt][i] = (__bf16)arow[i];
        }
        // Batch-load the NTW B fragments for this k-step
        v16b b[NTW];
        #pragma unroll
        for (int nt = 0; nt < NTW; ++nt) {
            const __bf16* bp = &ldsW[(nbase + nt * 16 + l16) * Kpad + k0 + (lhalf << 4)];
            #pragma unroll
            for (int i = 0; i < 16; ++i) b[nt][i] = bp[i];
        }
        // 2*NTW WMMAs; each B fragment feeds both row-tiles
        #pragma unroll
        for (int nt = 0; nt < NTW; ++nt) {
            acc[0][nt] = __builtin_amdgcn_wmma_f32_16x16x32_bf16(
                false, a[0], false, b[nt], (short)0, acc[0][nt], false, false);
            acc[1][nt] = __builtin_amdgcn_wmma_f32_16x16x32_bf16(
                false, a[1], false, b[nt], (short)0, acc[1][nt], false, false);
        }
    }

    // Epilogue. C/D layout: VGPR j -> M = j + 8*lhalf, N = l16 (+16*nt)
    const size_t rbase = (size_t)blockIdx.x * 128 + rowgrp * 32;
    #pragma unroll
    for (int mt = 0; mt < 2; ++mt) {
        #pragma unroll
        for (int nt = 0; nt < NTW; ++nt) {
            const int n = nbase + nt * 16 + l16;
            const float bv = BIAS ? bias[n] : 0.0f;
            #pragma unroll
            for (int j = 0; j < 8; ++j) {
                const int m = mt * 16 + j + (lhalf << 3);
                const size_t idx = (rbase + m) * (size_t)N + n;
                float v = silu_f(acc[mt][nt][j] + bv);
                if (MUL) v *= mul[idx];
                if (ADD) v += addv[idx];
                out[idx] = v;
            }
        }
    }
}

// ---------------------------------------------------------------------------
// rbf = (rbf0 @ W_rbf1[6,8]) @ W_rbf2[8,128] ; one thread per (edge, channel)
// ---------------------------------------------------------------------------
__global__ __launch_bounds__(256) void rbf_kernel(
    const float* __restrict__ rbf0, const float* __restrict__ W1,
    const float* __restrict__ W2, float* __restrict__ out)
{
    int gid = blockIdx.x * 256 + threadIdx.x;
    if (gid >= EDGES * HC) return;
    int e = gid >> 7, n = gid & 127;
    float r[NRAD];
    #pragma unroll
    for (int i = 0; i < NRAD; ++i) r[i] = rbf0[e * NRAD + i];
    float acc = 0.0f;
    #pragma unroll
    for (int b = 0; b < BASIS; ++b) {
        float s = 0.0f;
        #pragma unroll
        for (int i = 0; i < NRAD; ++i) s += r[i] * W1[i * BASIS + b];
        acc += s * W2[b * HC + n];
    }
    out[gid] = acc;
}

// ---------------------------------------------------------------------------
// Fused: sbf_t = (sbf@W_sbf1)@W_sbf2 ; msg = x_kj[idx_kj]*sbf_t ;
// agg[idx_ji] += msg  (global f32 atomics; agg is 32 MB -> L2-resident)
// One thread per triplet.
// ---------------------------------------------------------------------------
__global__ __launch_bounds__(256) void triplet_kernel(
    const float* __restrict__ sbf, const float* __restrict__ W1,
    const float* __restrict__ W2, const int* __restrict__ idx_kj,
    const int* __restrict__ idx_ji, const float* __restrict__ xkj,
    float* __restrict__ agg)
{
    __shared__ float sW1[NSPH * NRAD * BASIS]; // 42 x 8
    __shared__ float sW2[BASIS * INTC];        // 8 x 64
    int tid = threadIdx.x;
    for (int i = tid; i < NSPH * NRAD * BASIS; i += 256) sW1[i] = W1[i];
    for (int i = tid; i < BASIS * INTC; i += 256) sW2[i] = W2[i];
    __syncthreads();

    int t = blockIdx.x * 256 + tid;
    if (t >= TRIP) return;

    const float* srow = sbf + (size_t)t * (NSPH * NRAD);
    float s8[BASIS] = {};
    #pragma unroll
    for (int i = 0; i < NSPH * NRAD; ++i) {
        float sv = srow[i];
        #pragma unroll
        for (int b = 0; b < BASIS; ++b) s8[b] += sv * sW1[i * BASIS + b];
    }
    const int kj = idx_kj[t], ji = idx_ji[t];
    const float* xr = xkj + (size_t)kj * INTC;
    float* ar = agg + (size_t)ji * INTC;
    #pragma unroll 4
    for (int n = 0; n < INTC; ++n) {
        float v = 0.0f;
        #pragma unroll
        for (int b = 0; b < BASIS; ++b) v += s8[b] * sW2[b * INTC + n];
        atomicAdd(&ar[n], v * xr[n]);
    }
}

// ---------------------------------------------------------------------------
// e2 = (rbf0 @ W_rbf[6,128]) * e1 ; one thread per (edge, channel)
// ---------------------------------------------------------------------------
__global__ __launch_bounds__(256) void e2_kernel(
    const float* __restrict__ rbf0, const float* __restrict__ Wrbf,
    const float* __restrict__ e1, float* __restrict__ e2)
{
    int gid = blockIdx.x * 256 + threadIdx.x;
    if (gid >= EDGES * HC) return;
    int e = gid >> 7, n = gid & 127;
    float r = 0.0f;
    #pragma unroll
    for (int i = 0; i < NRAD; ++i) r += rbf0[e * NRAD + i] * Wrbf[i * HC + n];
    e2[gid] = r * e1[gid];
}

extern "C" void kernel_launch(void* const* d_in, const int* in_sizes, int n_in,
                              void* d_out, int out_size, void* d_ws, size_t ws_size,
                              hipStream_t stream)
{
    (void)in_sizes; (void)n_in; (void)out_size; (void)ws_size;

    const float* x1      = (const float*)d_in[0];
    const float* rbf0    = (const float*)d_in[1];
    const float* sbf     = (const float*)d_in[2];
    const int*   idx_kj  = (const int*)d_in[3];
    const int*   idx_ji  = (const int*)d_in[4];
    const float* lin_ji_w = (const float*)d_in[5];
    const float* lin_ji_b = (const float*)d_in[6];
    const float* lin_kj_w = (const float*)d_in[7];
    const float* lin_kj_b = (const float*)d_in[8];
    const float* W_rbf1   = (const float*)d_in[9];
    const float* W_rbf2   = (const float*)d_in[10];
    const float* W_sbf1   = (const float*)d_in[11];
    const float* W_sbf2   = (const float*)d_in[12];
    const float* W_down   = (const float*)d_in[13];
    const float* W_up     = (const float*)d_in[14];
    const float* lin_w    = (const float*)d_in[15];
    const float* lin_b    = (const float*)d_in[16];
    const float* W_rbf    = (const float*)d_in[17];
    const float* b0l1_w   = (const float*)d_in[18];
    const float* b0l1_b   = (const float*)d_in[19];
    const float* b0l2_w   = (const float*)d_in[20];
    const float* b0l2_b   = (const float*)d_in[21];
    const float* a0l1_w   = (const float*)d_in[22];
    const float* a0l1_b   = (const float*)d_in[23];
    const float* a0l2_w   = (const float*)d_in[24];
    const float* a0l2_b   = (const float*)d_in[25];
    const float* a1l1_w   = (const float*)d_in[26];
    const float* a1l1_b   = (const float*)d_in[27];
    const float* a1l2_w   = (const float*)d_in[28];
    const float* a1l2_b   = (const float*)d_in[29];

    float* e1 = (float*)d_out;
    float* e2 = e1 + (size_t)EDGES * HC;

    // Workspace layout (3 x [E,128] fp32 = 192 MB), with liveness-based reuse:
    float* XJI  = (float*)d_ws;                 // [E,128] x_ji         (live: G1 -> G4)
    float* TKJ  = XJI + (size_t)EDGES * HC;     // [E,128] t_kj         (G2 -> G3), reused as Hbuf
    float* RBF  = TKJ + (size_t)EDGES * HC;     // [E,128] rbf          (rbf -> G2)
    float* XKJI = RBF;                          // [E,64]  x_kj(int)    (G3 -> triplet) reuses RBF
    float* AGG  = RBF + (size_t)EDGES * INTC;   // [E,64]  segment sum  (triplet -> G4)
    float* Hbuf = TKJ;                          // residual hidden (after G3, TKJ is dead)

    dim3 blk(256);
    dim3 gEH((EDGES * HC) / 256);   // 65536 blocks for per-(e,n) kernels
    dim3 gg(EDGES / 128);           // 1024 blocks for GEMMs
    dim3 gT(TRIP / 256);            // 4096 blocks for triplets

    // rbf = (rbf0@W_rbf1)@W_rbf2
    rbf_kernel<<<gEH, blk, 0, stream>>>(rbf0, W_rbf1, W_rbf2, RBF);

    // x_ji = silu(x1@W_ji + b)
    wmma_gemm_kernel<HC, HC, true, false, false>
        <<<gg, blk, 0, stream>>>(x1, lin_ji_w, lin_ji_b, nullptr, nullptr, XJI);
    // t_kj = silu(x1@W_kj + b) * rbf
    wmma_gemm_kernel<HC, HC, true, true, false>
        <<<gg, blk, 0, stream>>>(x1, lin_kj_w, lin_kj_b, RBF, nullptr, TKJ);
    // x_kj_int = silu(t_kj @ W_down)   [E,64]
    wmma_gemm_kernel<HC, INTC, false, false, false>
        <<<gg, blk, 0, stream>>>(TKJ, W_down, nullptr, nullptr, nullptr, XKJI);

    hipMemsetAsync(AGG, 0, (size_t)EDGES * INTC * sizeof(float), stream);
    // fused sbf transform + gather + scatter-sum
    triplet_kernel<<<gT, blk, 0, stream>>>(sbf, W_sbf1, W_sbf2, idx_kj, idx_ji, XKJI, AGG);

    // e1 = x_ji + silu(agg @ W_up)
    wmma_gemm_kernel<INTC, HC, false, false, true>
        <<<gg, blk, 0, stream>>>(AGG, W_up, nullptr, nullptr, XJI, e1);

    // before-skip residual
    wmma_gemm_kernel<HC, HC, true, false, false>
        <<<gg, blk, 0, stream>>>(e1, b0l1_w, b0l1_b, nullptr, nullptr, Hbuf);
    wmma_gemm_kernel<HC, HC, true, false, true>
        <<<gg, blk, 0, stream>>>(Hbuf, b0l2_w, b0l2_b, nullptr, e1, e1);

    // e1 = silu(e1@W + b) + x1
    wmma_gemm_kernel<HC, HC, true, false, true>
        <<<gg, blk, 0, stream>>>(e1, lin_w, lin_b, nullptr, x1, e1);

    // after-skip residuals x2
    wmma_gemm_kernel<HC, HC, true, false, false>
        <<<gg, blk, 0, stream>>>(e1, a0l1_w, a0l1_b, nullptr, nullptr, Hbuf);
    wmma_gemm_kernel<HC, HC, true, false, true>
        <<<gg, blk, 0, stream>>>(Hbuf, a0l2_w, a0l2_b, nullptr, e1, e1);
    wmma_gemm_kernel<HC, HC, true, false, false>
        <<<gg, blk, 0, stream>>>(e1, a1l1_w, a1l1_b, nullptr, nullptr, Hbuf);
    wmma_gemm_kernel<HC, HC, true, false, true>
        <<<gg, blk, 0, stream>>>(Hbuf, a1l2_w, a1l2_b, nullptr, e1, e1);

    // e2 = (rbf0@W_rbf) * e1
    e2_kernel<<<gEH, blk, 0, stream>>>(rbf0, W_rbf, e1, e2);
}
